// Mapping_32959579029666
// MI455X (gfx1250) — matrix operand order repaired and checked by reading, not compiled, not run on hardware
//
#include <hip/hip_runtime.h>
#include <stdint.h>

// ---------- problem constants ----------
#define BATCH 8
#define N1    2048
#define N2    4096
#define D1    64
#define D2    64
#define CIN   131           // D1 + D2 + 3
#define KNN   16
#define HID   128
#define LEAKY 0.1f

typedef __attribute__((ext_vector_type(8)))  float  v8f;
typedef __attribute__((ext_vector_type(16))) __bf16 v16bf;

union FragB16 {
    v16bf    v;
    uint16_t u[16];
    uint4    q[2];
};

__device__ __forceinline__ uint16_t f32_bf16(float f) {
    return __builtin_bit_cast(uint16_t, (__bf16)f);      // native v_cvt path
}

__device__ __forceinline__ float bf16_f32(uint16_t u) {
    return (float)__builtin_bit_cast(__bf16, u);         // fpext = shl 16
}

__device__ __forceinline__ uint32_t pack_bf16x2(float lo, float hi) {
    // pairwise pattern -> v_cvt_pk_bf16_f32 when available
    return (uint32_t)f32_bf16(lo) | ((uint32_t)f32_bf16(hi) << 16);
}

__device__ __forceinline__ float leaky(float x) {
    return fmaxf(x, LEAKY * x);            // single v_max_f32 after mul
}

// ============================================================================
// Kernel 0: pack fp32 weights [Cout=128, Cin] into bf16 WMMA B-fragments.
// Fragment order: [chunk][tile][lane][16 elems] — per-lane contiguous 32B so
// the consumer loads each fragment as two b128s.
// B-operand layout (16-bit, 32x16): lanes 0-15 hold K=0..7,16..23 of col=lane,
// lanes 16-31 hold K=8..15,24..31 of col=lane-16; 2 K-values per VGPR.
// koff selects a column-slice of the source weight (for the w0 split).
// ============================================================================
__global__ void pack_weights_kernel(const float* __restrict__ src,
                                    uint16_t* __restrict__ dst,
                                    int cin, int koff, int total) {
    int gid = blockIdx.x * blockDim.x + threadIdx.x;
    if (gid >= total) return;
    int e    = gid & 15;
    int lane = (gid >> 4) & 31;
    int t    = (gid >> 9) & 7;
    int c    = gid >> 12;
    int col  = t * 16 + (lane & 15);
    int k    = koff + c * 32 + ((lane >> 4) << 3) + (e & 7) + ((e >> 3) << 4);
    float v  = (k < cin) ? src[col * cin + k] : 0.0f;
    dst[gid] = f32_bf16(v);
}

// ============================================================================
// Kernel 1: exact K=16 nearest neighbors. One thread per query, xyz2 tiled
// through LDS, register insertion-sort (fully unrolled -> stays in VGPRs).
// ============================================================================
__global__ __launch_bounds__(128) void knn_kernel(const float* __restrict__ xyz1,
                                                  const float* __restrict__ xyz2,
                                                  int* __restrict__ knnIdx) {
    const int b = blockIdx.x >> 4;                       // 16 blocks per batch
    const int n = ((blockIdx.x & 15) << 7) + threadIdx.x;

    const float qx = xyz1[((size_t)b * 3 + 0) * N1 + n];
    const float qy = xyz1[((size_t)b * 3 + 1) * N1 + n];
    const float qz = xyz1[((size_t)b * 3 + 2) * N1 + n];

    __shared__ float sx[256], sy[256], sz[256];

    float dist[KNN];
    int   idx[KNN];
#pragma unroll
    for (int j = 0; j < KNN; ++j) { dist[j] = 3.4e38f; idx[j] = 0; }

    for (int t0 = 0; t0 < N2; t0 += 256) {
        __syncthreads();
        for (int i = threadIdx.x; i < 256; i += 128) {
            sx[i] = xyz2[((size_t)b * 3 + 0) * N2 + t0 + i];
            sy[i] = xyz2[((size_t)b * 3 + 1) * N2 + t0 + i];
            sz[i] = xyz2[((size_t)b * 3 + 2) * N2 + t0 + i];
        }
        __syncthreads();
#pragma unroll 1
        for (int i = 0; i < 256; ++i) {
            float dx = sx[i] - qx, dy = sy[i] - qy, dz = sz[i] - qz;
            float d  = dx * dx + dy * dy + dz * dz;
            if (d < dist[KNN - 1]) {           // candidate beats current worst
                float cd = d; int ci = t0 + i;
#pragma unroll
                for (int j = 0; j < KNN; ++j) { // sorted-ascending insertion
                    bool lt = cd < dist[j];
                    float td = dist[j]; int ti = idx[j];
                    dist[j] = lt ? cd : td;  idx[j] = lt ? ci : ti;
                    cd      = lt ? td : cd;  ci     = lt ? ti : ci;
                }
            }
        }
    }
#pragma unroll
    for (int j = 0; j < KNN; ++j)
        knnIdx[((size_t)(b * N1 + n)) * KNN + j] = idx[j];
}

// ============================================================================
// Kernel 2: fused gather + MLP, all WMMA.
// One wave == 16 query points.
//   Pre-pass : P1[16q x 64] @ w0a (+b0 seed) -> p1c[16 x 128] in LDS (bf16).
//              (grouped_p1 is broadcast over K and added pre-activation, so
//               its layer-0 contribution is per-query rank-1 -> compute once.)
//   Per query: feat = p2||dir [16 x 96] bf16 staged in LDS (paired b32
//              stores); layer0 = 3x8 WMMA seeded with splat(p1c[q]);
//              layer1 = 4x8 WMMA seeded with b1; maxpool via 8-row reduce +
//              ds_swizzle SWAPX16.
//   Epilogue : pooled[16q x 128] @ w2 = 4x8 WMMA, leaky, store.
// ============================================================================
#define WAVES     4
#define FEAT_COLS 96
#define FEAT_ELTS (16 * FEAT_COLS)
#define H_ELTS    (16 * 128)
#define POOL_ELTS (16 * 128)
#define P1C_ELTS  (16 * 128)
#define WAVE_LDS  (FEAT_ELTS + H_ELTS + POOL_ELTS + P1C_ELTS)  // 7680 u16 = 15 KiB

__global__ __launch_bounds__(128) void fused_mlp_kernel(
    const float* __restrict__ xyz1, const float* __restrict__ xyz2,
    const float* __restrict__ p1,   const float* __restrict__ p2,
    const int*   __restrict__ knn,
    const uint16_t* __restrict__ w0a, const uint16_t* __restrict__ w0b,
    const float* __restrict__ b0,
    const uint16_t* __restrict__ w1p, const float* __restrict__ b1,
    const uint16_t* __restrict__ w2p,
    float* __restrict__ out) {

    __shared__ uint16_t smem[WAVES * WAVE_LDS];

    const int lane = threadIdx.x & 31;
    const int wid  = threadIdx.x >> 5;
    const int tile = blockIdx.x * WAVES + wid;       // 1024 tiles total
    const int b    = tile >> 7;                      // N1/16 = 128 tiles/batch
    const int nb   = (tile & 127) << 4;              // query base

    uint16_t* featL = smem + wid * WAVE_LDS;
    uint16_t* hL    = featL + FEAT_ELTS;
    uint16_t* poolL = hL + H_ELTS;
    uint16_t* p1cL  = poolL + POOL_ELTS;

    const int half = lane >> 4;                      // A/C fragment half
    const int l15  = lane & 15;
    const int row  = l15;

    const float* p1b = p1 + (size_t)b * D1 * N1;
    const float* p2b = p2 + (size_t)b * D2 * N2;

    v8f acc[8];

    // ---- pre-pass: p1 contribution for all 16 queries, bias folded in ----
    {
#pragma unroll
        for (int t = 0; t < 8; ++t) {
            const float bb = b0[t * 16 + l15];
#pragma unroll
            for (int e = 0; e < 8; ++e) acc[t][e] = bb;
        }
        const int nq = nb + l15;                     // A row = query id
#pragma unroll
        for (int c = 0; c < 2; ++c) {
            FragB16 a;
            const int kb = c * 32 + half * 8;
#pragma unroll
            for (int j = 0; j < 8; ++j) {
                a.u[j]     = f32_bf16(p1b[(kb + j) * N1 + nq]);
                a.u[8 + j] = f32_bf16(p1b[(kb + 16 + j) * N1 + nq]);
            }
#pragma unroll
            for (int t = 0; t < 8; ++t) {
                FragB16 w;
                const uint4* wp = (const uint4*)(w0a + (((c * 8 + t) * 32 + lane) << 4));
                w.q[0] = wp[0]; w.q[1] = wp[1];
                acc[t] = __builtin_amdgcn_wmma_f32_16x16x32_bf16(
                    false, a.v, false, w.v, (short)0, acc[t], false, false);
            }
        }
        // store pre-activation p1 contribution rows (row = query)
#pragma unroll
        for (int t = 0; t < 8; ++t)
#pragma unroll
            for (int e = 0; e < 8; ++e)
                p1cL[(e + half * 8) * 128 + t * 16 + l15] = f32_bf16(acc[t][e]);
    }
    // zero-pad feat cols 68..95 once (constant across queries)
    if (half == 1) {
        uint32_t* frow32 = (uint32_t*)(featL + row * FEAT_COLS);
#pragma unroll
        for (int jp = 34; jp < 48; ++jp) frow32[jp] = 0;
    }
    asm volatile("s_wait_dscnt 0" ::: "memory");

#pragma unroll 1
    for (int q = 0; q < 16; ++q) {
        const int n = nb + q;
        __builtin_prefetch(&knn[((size_t)(b * N1 + n + 1)) * KNN], 0, 0);

        // ---- stage feat[16 x 96] = p2-gather || direction, bf16 pairs ----
        const int nidx = knn[((size_t)(b * N1 + n)) * KNN + row];
        uint32_t* frow32 = (uint32_t*)(featL + row * FEAT_COLS);
        if (half == 0) {
#pragma unroll 1
            for (int jp = 0; jp < 17; ++jp) {        // cols 0..33
                float lo = p2b[(2 * jp + 0) * N2 + nidx];
                float hi = p2b[(2 * jp + 1) * N2 + nidx];
                frow32[jp] = pack_bf16x2(lo, hi);
            }
        } else {
#pragma unroll 1
            for (int jp = 17; jp < 32; ++jp) {       // cols 34..63
                float lo = p2b[(2 * jp + 0) * N2 + nidx];
                float hi = p2b[(2 * jp + 1) * N2 + nidx];
                frow32[jp] = pack_bf16x2(lo, hi);
            }
            float d0 = xyz2[((size_t)b * 3 + 0) * N2 + nidx]
                     - xyz1[((size_t)b * 3 + 0) * N1 + n];
            float d1 = xyz2[((size_t)b * 3 + 1) * N2 + nidx]
                     - xyz1[((size_t)b * 3 + 1) * N1 + n];
            float d2 = xyz2[((size_t)b * 3 + 2) * N2 + nidx]
                     - xyz1[((size_t)b * 3 + 2) * N1 + n];
            frow32[32] = pack_bf16x2(d0, d1);        // cols 64,65
            frow32[33] = pack_bf16x2(d2, 0.0f);      // cols 66,67(pad)
        }
        asm volatile("s_wait_dscnt 0" ::: "memory");  // wave-private LDS ready

        // ---- layer 0: seed = p1c[q] splat, add p2||dir @ w0b (3 chunks) ----
#pragma unroll
        for (int t = 0; t < 8; ++t) {
            const float base = bf16_f32(p1cL[q * 128 + t * 16 + l15]);
#pragma unroll
            for (int e = 0; e < 8; ++e) acc[t][e] = base;
        }
#pragma unroll
        for (int c = 0; c < 3; ++c) {
            FragB16 a;
            const uint16_t* fr = featL + row * FEAT_COLS;
            const int kb = c * 32 + half * 8;         // A layout: 2 runs of 8
#pragma unroll
            for (int j = 0; j < 8; ++j) { a.u[j] = fr[kb + j]; a.u[8 + j] = fr[kb + 16 + j]; }
#pragma unroll
            for (int t = 0; t < 8; ++t) {
                FragB16 w;
                const uint4* wp = (const uint4*)(w0b + (((c * 8 + t) * 32 + lane) << 4));
                w.q[0] = wp[0]; w.q[1] = wp[1];
                acc[t] = __builtin_amdgcn_wmma_f32_16x16x32_bf16(
                    false, a.v, false, w.v, (short)0, acc[t], false, false);
            }
        }
        // leaky, write h[16x128] bf16 (C layout -> row-major LDS)
#pragma unroll
        for (int t = 0; t < 8; ++t)
#pragma unroll
            for (int e = 0; e < 8; ++e)
                hL[(e + half * 8) * 128 + t * 16 + l15] = f32_bf16(leaky(acc[t][e]));
        asm volatile("s_wait_dscnt 0" ::: "memory");

        // ---- layer 1: [16x128] @ [128x128], bias seeded ----
#pragma unroll
        for (int t = 0; t < 8; ++t) {
            const float bb = b1[t * 16 + l15];
#pragma unroll
            for (int e = 0; e < 8; ++e) acc[t][e] = bb;
        }
#pragma unroll
        for (int c = 0; c < 4; ++c) {
            FragB16 a;
            const uint16_t* fr = hL + row * 128;
            const int kb = c * 32 + half * 8;
#pragma unroll
            for (int j = 0; j < 8; ++j) { a.u[j] = fr[kb + j]; a.u[8 + j] = fr[kb + 16 + j]; }
#pragma unroll
            for (int t = 0; t < 8; ++t) {
                FragB16 w;
                const uint4* wp = (const uint4*)(w1p + (((c * 8 + t) * 32 + lane) << 4));
                w.q[0] = wp[0]; w.q[1] = wp[1];
                acc[t] = __builtin_amdgcn_wmma_f32_16x16x32_bf16(
                    false, a.v, false, w.v, (short)0, acc[t], false, false);
            }
        }
        // leaky + maxpool over 16 neighbors (8 local rows + SWAPX16)
#pragma unroll
        for (int t = 0; t < 8; ++t) {
            float m = -3.4e38f;
#pragma unroll
            for (int e = 0; e < 8; ++e)
                m = fmaxf(m, leaky(acc[t][e]));
            int other = __builtin_amdgcn_ds_swizzle(__float_as_int(m), 0x401F); // xor 16
            m = fmaxf(m, __int_as_float(other));
            if (half == 0)
                poolL[q * 128 + t * 16 + l15] = f32_bf16(m);
        }
    }
    asm volatile("s_wait_dscnt 0" ::: "memory");

    // ---- layer 2: pooled [16 queries x 128] @ [128x128], no bias ----
#pragma unroll
    for (int t = 0; t < 8; ++t)
#pragma unroll
        for (int e = 0; e < 8; ++e) acc[t][e] = 0.0f;
#pragma unroll
    for (int c = 0; c < 4; ++c) {
        FragB16 a;
        const uint16_t* fr = poolL + l15 * 128;       // row = query id
        const int kb = c * 32 + half * 8;
#pragma unroll
        for (int j = 0; j < 8; ++j) { a.u[j] = fr[kb + j]; a.u[8 + j] = fr[kb + 16 + j]; }
#pragma unroll
        for (int t = 0; t < 8; ++t) {
            FragB16 w;
            const uint4* wp = (const uint4*)(w2p + (((c * 8 + t) * 32 + lane) << 4));
            w.q[0] = wp[0]; w.q[1] = wp[1];
            acc[t] = __builtin_amdgcn_wmma_f32_16x16x32_bf16(
                false, a.v, false, w.v, (short)0, acc[t], false, false);
        }
    }
    // leaky + store out[b, 128, N1]; C layout: M=query=e+8*half, N=t*16+l15
#pragma unroll
    for (int t = 0; t < 8; ++t) {
        const int ch = t * 16 + l15;
#pragma unroll
        for (int e = 0; e < 8; ++e) {
            float x = leaky(acc[t][e]);
            out[((size_t)b * HID + ch) * N1 + nb + (e + half * 8)] = x;
        }
    }
}

// ============================================================================
extern "C" void kernel_launch(void* const* d_in, const int* in_sizes, int n_in,
                              void* d_out, int out_size, void* d_ws, size_t ws_size,
                              hipStream_t stream) {
    const float* xyz1 = (const float*)d_in[0];
    const float* xyz2 = (const float*)d_in[1];
    const float* pts1 = (const float*)d_in[2];
    const float* pts2 = (const float*)d_in[3];
    const float* w0   = (const float*)d_in[4];
    const float* b0   = (const float*)d_in[5];
    const float* w1   = (const float*)d_in[6];
    const float* b1   = (const float*)d_in[7];
    const float* w2   = (const float*)d_in[8];
    float* out        = (float*)d_out;

    // workspace layout (bytes)
    char* ws = (char*)d_ws;
    int*      wsIdx = (int*)ws;                              // 8*2048*16*4 = 1 MiB (2 MiB reserved)
    uint16_t* w0a = (uint16_t*)(ws + 2097152);               // 2 chunks: 16 KiB
    uint16_t* w0b = (uint16_t*)(ws + 2097152 + 16384);       // 3 chunks: 24 KiB
    uint16_t* w1p = (uint16_t*)(ws + 2097152 + 16384 + 24576);           // 32 KiB
    uint16_t* w2p = (uint16_t*)(ws + 2097152 + 16384 + 24576 + 32768);   // 32 KiB

    // 0) pack weights into WMMA bf16 fragments (w0 split at k=64)
    pack_weights_kernel<<<32, 256, 0, stream>>>(w0, w0a, CIN, 0,  2 * 8 * 32 * 16);
    pack_weights_kernel<<<48, 256, 0, stream>>>(w0, w0b, CIN, 64, 3 * 8 * 32 * 16);
    pack_weights_kernel<<<64, 256, 0, stream>>>(w1, w1p, HID, 0,  4 * 8 * 32 * 16);
    pack_weights_kernel<<<64, 256, 0, stream>>>(w2, w2p, HID, 0,  4 * 8 * 32 * 16);

    // 1) exact top-16 KNN
    knn_kernel<<<BATCH * (N1 / 128), 128, 0, stream>>>(xyz1, xyz2, wsIdx);

    // 2) fused gather + MLP (WMMA)
    fused_mlp_kernel<<<(BATCH * N1 / 16) / WAVES, 32 * WAVES, 0, stream>>>(
        xyz1, xyz2, pts1, pts2, wsIdx, w0a, w0b, b0, w1p, b1, w2p, out);
}
